// SparseEncoder_87754771792445
// MI455X (gfx1250) — matrix operand order repaired and checked by reading, not compiled, not run on hardware
//
#include <hip/hip_runtime.h>
#include <hip/hip_bf16.h>

// ---------------------------------------------------------------------------
// Types for CDNA5 WMMA (wave32): V_WMMA_F32_16X16X32_F16
// ---------------------------------------------------------------------------
typedef __attribute__((ext_vector_type(16))) _Float16 v16h;
typedef __attribute__((ext_vector_type(8)))  _Float16 v8h;
typedef __attribute__((ext_vector_type(8)))  float    v8f;

__device__ __forceinline__ _Float16 f2h(float f) { return (_Float16)f; }

__device__ __forceinline__ v16h cat8(v8h lo, v8h hi) {
    return __builtin_shufflevector(lo, hi, 0,1,2,3,4,5,6,7,8,9,10,11,12,13,14,15);
}

// LDS byte offset of a generic pointer to __shared__ (low 32 bits of the address)
__device__ __forceinline__ unsigned lds_off(const void* p) {
    return (unsigned)(uintptr_t)p;
}

// Async copy: 16B / 4B from global to LDS, tracked on ASYNCcnt.
__device__ __forceinline__ void async_b128(unsigned lds, const float* g) {
    asm volatile("global_load_async_to_lds_b128 %0, %1, off"
                 :: "v"(lds), "v"((unsigned long long)(uintptr_t)g) : "memory");
}
__device__ __forceinline__ void async_b32(unsigned lds, const float* g) {
    asm volatile("global_load_async_to_lds_b32 %0, %1, off"
                 :: "v"(lds), "v"((unsigned long long)(uintptr_t)g) : "memory");
}

// Pack fragments from fp32 LDS tiles (A 16x32 / B 32x16, contiguous k = hi*16 + s, same for A and B).
__device__ __forceinline__ v16h packA(const float* p) {   // K hi*16..+15 contiguous
    union { v16h v; _Float16 e[16]; } u;
#pragma unroll
    for (int i = 0; i < 8; ++i) u.e[i]     = f2h(p[i]);
#pragma unroll
    for (int i = 0; i < 8; ++i) u.e[8 + i] = f2h(p[8 + i]);
    return u.v;
}
__device__ __forceinline__ v16h packB(const float* p) {   // K hi*16..+15 contiguous
    union { v16h v; _Float16 e[16]; } u;
#pragma unroll
    for (int i = 0; i < 16; ++i) u.e[i] = f2h(p[i]);
    return u.v;
}

// ---------------------------------------------------------------------------
// GEMM: C[M,N] = A[M,K] @ W + bias
//   w_is_nk == 0 : W stored [K,N];  w_is_nk == 1 : W stored [N,K] (C = A@W^T)
// 128x64 macro-tile per 256-thread block; 8 waves, each a 32x32 tile
// (2 A-frags x 2 B-frags -> 4 WMMAs per K-step of 32).
// Double-buffered LDS fed by global_load_async_to_lds_* / s_wait_asynccnt.
// Per-thread async issues per stage: A: 4 x b128; W: 2 x b128 (nk) or 8 x b32
// (scatter-transpose of [K,N] weights). fp32 tiles, cvt_pk at fragment build.
// ---------------------------------------------------------------------------
#define TSTR 36   // fp32 LDS row stride: 144 B, keeps 16B alignment

__global__ void __launch_bounds__(256)
gemm_f16_wmma(const float* __restrict__ A, const float* __restrict__ W,
              const float* __restrict__ bias, float* __restrict__ C,
              int M, int N, int K, int w_is_nk)
{
    __shared__ __align__(16) float Asb[2][128 * TSTR];
    __shared__ __align__(16) float Wsb[2][64 * TSTR];

    const int tid  = threadIdx.x;
    const int lane = tid & 31;
    const int wave = tid >> 5;
    const int wm   = wave & 3;    // M band (32 rows each)
    const int wn   = wave >> 2;   // N band (32 cols each)
    const int bm   = blockIdx.y * 128;
    const int bn   = blockIdx.x * 64;
    const int mloc = lane & 15;
    const int hi   = lane >> 4;

    v8f acc00 = {}, acc01 = {}, acc10 = {}, acc11 = {};

    auto stage = [&](int k0, int buf) {
        // A tile 128 x 32 : 1024 16B-chunks, 4 per thread
#pragma unroll
        for (int i = 0; i < 4; ++i) {
            const int ch  = tid + (i << 8);
            const int row = ch >> 3;
            const int col = (ch & 7) << 2;
            async_b128(lds_off(&Asb[buf][row * TSTR + col]),
                       A + (size_t)(bm + row) * K + k0 + col);
        }
        if (w_is_nk) {
            // W[N,K]: rows are already the transposed layout -> 16B chunks
#pragma unroll
            for (int i = 0; i < 2; ++i) {
                const int ch  = tid + (i << 8);
                const int n   = ch >> 3;
                const int col = (ch & 7) << 2;
                async_b128(lds_off(&Wsb[buf][n * TSTR + col]),
                           W + (size_t)(bn + n) * K + k0 + col);
            }
        } else {
            // W[K,N]: element-wise async b32 scatter into Wt[n][k]
#pragma unroll
            for (int i = 0; i < 8; ++i) {
                const int e = tid + (i << 8);     // 0..2047
                const int k = e >> 6;
                const int n = e & 63;
                async_b32(lds_off(&Wsb[buf][n * TSTR + k]),
                          W + (size_t)(k0 + k) * N + bn + n);
            }
        }
    };

    stage(0, 0);
    const int nT = K >> 5;
    for (int t = 0; t < nT; ++t) {
        const int buf = t & 1;
        if (t + 1 < nT) {
            stage((t + 1) << 5, buf ^ 1);
            // drain stage t (<= issues of one stage still outstanding)
            if (w_is_nk) asm volatile("s_wait_asynccnt 6"  ::: "memory");
            else         asm volatile("s_wait_asynccnt 12" ::: "memory");
        } else {
            asm volatile("s_wait_asynccnt 0" ::: "memory");
        }
        __syncthreads();

        const float* Ab = &Asb[buf][0];
        const float* Wb = &Wsb[buf][0];
        v16h a0 = packA(&Ab[(wm * 32 + mloc) * TSTR + hi * 16]);
        v16h a1 = packA(&Ab[(wm * 32 + 16 + mloc) * TSTR + hi * 16]);
        v16h b0 = packB(&Wb[(wn * 32 + mloc) * TSTR + hi * 16]);
        v16h b1 = packB(&Wb[(wn * 32 + 16 + mloc) * TSTR + hi * 16]);

        acc00 = __builtin_amdgcn_wmma_f32_16x16x32_f16(false, a0, false, b0,
                                                       (short)0, acc00, false, false);
        acc01 = __builtin_amdgcn_wmma_f32_16x16x32_f16(false, a0, false, b1,
                                                       (short)0, acc01, false, false);
        acc10 = __builtin_amdgcn_wmma_f32_16x16x32_f16(false, a1, false, b0,
                                                       (short)0, acc10, false, false);
        acc11 = __builtin_amdgcn_wmma_f32_16x16x32_f16(false, a1, false, b1,
                                                       (short)0, acc11, false, false);
        __syncthreads();   // protect buf before it is overwritten by stage t+2
    }

    // epilogue: C/D layout — lane holds col (lane&15), rows hi*8 + r
    const int cg0 = bn + wn * 32 + mloc;
    const int cg1 = cg0 + 16;
    const float bv0 = bias ? bias[cg0] : 0.0f;
    const float bv1 = bias ? bias[cg1] : 0.0f;
#pragma unroll
    for (int r = 0; r < 8; ++r) {
        const int row0 = bm + wm * 32 + hi * 8 + r;
        const int row1 = row0 + 16;
        C[(size_t)row0 * N + cg0] = acc00[r] + bv0;
        C[(size_t)row0 * N + cg1] = acc01[r] + bv1;
        C[(size_t)row1 * N + cg0] = acc10[r] + bv0;
        C[(size_t)row1 * N + cg1] = acc11[r] + bv1;
    }
}

// ---------------------------------------------------------------------------
// Row-wise LayerNorm + exact GELU, in place. One block per row.
// ---------------------------------------------------------------------------
__device__ __forceinline__ float block_reduce_sum(float v, float* red)
{
    const int lane = threadIdx.x & 31, wave = threadIdx.x >> 5;
#pragma unroll
    for (int off = 16; off; off >>= 1) v += __shfl_xor(v, off, 32);
    if (lane == 0) red[wave] = v;
    __syncthreads();
    if (wave == 0) {
        v = (lane < 8) ? red[lane] : 0.0f;
#pragma unroll
        for (int off = 4; off; off >>= 1) v += __shfl_xor(v, off, 32);
        if (lane == 0) red[0] = v;
    }
    __syncthreads();
    float r = red[0];
    __syncthreads();
    return r;
}

__global__ void __launch_bounds__(256)
ln_gelu_kernel(float* __restrict__ h, const float* __restrict__ g,
               const float* __restrict__ b, int D2)
{
    __shared__ float red[8];
    float* p = h + (size_t)blockIdx.x * D2;

    float s = 0.0f;
    for (int c = threadIdx.x; c < D2; c += 256) s += p[c];
    const float mu = block_reduce_sum(s, red) / (float)D2;

    float vs = 0.0f;
    for (int c = threadIdx.x; c < D2; c += 256) {
        const float d = p[c] - mu;
        vs += d * d;
    }
    const float var = block_reduce_sum(vs, red) / (float)D2;
    const float inv = rsqrtf(var + 1e-5f);

    for (int c = threadIdx.x; c < D2; c += 256) {
        const float y = (p[c] - mu) * inv * g[c] + b[c];
        p[c] = 0.5f * y * (1.0f + erff(y * 0.70710678118654752f));
    }
}

// ---------------------------------------------------------------------------
// Cross-attention over the fixed 2048-entry codebook KV.
// Block = (head, 16-query tile). Dynamic LDS: qs 16x128 f16 | sc 16x2048 f32 |
// ps 16x2048 f16 | rsum. scores & ctx both via WMMA; softmax with wave32
// shuffles; 1/rowsum folded into the ctx store.
// ---------------------------------------------------------------------------
#define ATTN_SMEM_BYTES (4096 + 131072 + 65536 + 64)

__global__ void __launch_bounds__(256)
attn_kernel(const float* __restrict__ Q, const float* __restrict__ Km,
            const float* __restrict__ Vm, float* __restrict__ ctx)
{
    extern __shared__ char smem[];
    _Float16* qs   = (_Float16*)smem;                      // 16 x 128
    float*    sc   = (float*)(smem + 4096);                // 16 x 2048
    _Float16* ps   = (_Float16*)(smem + 4096 + 131072);    // 16 x 2048
    float*    rsum = (float*)(smem + 4096 + 131072 + 65536);

    const int tid  = threadIdx.x;
    const int lane = tid & 31;
    const int wave = tid >> 5;
    const int h    = blockIdx.x;        // head 0..7
    const int row0 = blockIdx.y * 16;   // token tile
    const int mloc = lane & 15;
    const int hi   = lane >> 4;
    const float scale = 0.08838834764831845f;  // 1/sqrt(128)

    // stage q tile (scaled) as f16
    {
        const int r = tid >> 4, c = (tid & 15) << 3;
        const float* src = Q + (size_t)(row0 + r) * 1024 + h * 128 + c;
        float f[8];
        *(float4*)(f)     = *(const float4*)(src);
        *(float4*)(f + 4) = *(const float4*)(src + 4);
#pragma unroll
        for (int i = 0; i < 8; ++i) qs[r * 128 + c + i] = f2h(f[i] * scale);
    }
    __syncthreads();

    // ---- phase 1: score tiles ----
    for (int j = wave; j < 128; j += 8) {
        const int kb = j * 16;
        v8f acc = {};
#pragma unroll
        for (int kc = 0; kc < 4; ++kc) {
            const v8h* pa = (const v8h*)&qs[mloc * 128 + kc * 32 + hi * 16];
            v16h a = cat8(pa[0], pa[1]);
            union { v16h v; _Float16 e[16]; } bu;
            const float* kp = Km + (size_t)(kb + mloc) * 1024 + h * 128 + kc * 32 + hi * 16;
#pragma unroll
            for (int i = 0; i < 16; ++i) bu.e[i] = f2h(kp[i]);
            acc = __builtin_amdgcn_wmma_f32_16x16x32_f16(false, a, false, bu.v,
                                                         (short)0, acc, false, false);
        }
#pragma unroll
        for (int r = 0; r < 8; ++r)
            sc[(hi * 8 + r) * 2048 + kb + mloc] = acc[r];
    }
    __syncthreads();

    // ---- phase 2: softmax (2 rows per wave, wave32 shuffles) ----
#pragma unroll
    for (int rr = 0; rr < 2; ++rr) {
        const int row = wave * 2 + rr;
        float mx = -3.0e38f;
        for (int c = lane; c < 2048; c += 32) mx = fmaxf(mx, sc[row * 2048 + c]);
#pragma unroll
        for (int off = 16; off; off >>= 1) mx = fmaxf(mx, __shfl_xor(mx, off, 32));
        float sum = 0.0f;
        for (int c = lane; c < 2048; c += 32) {
            const float e = __expf(sc[row * 2048 + c] - mx);
            sum += e;
            ps[row * 2048 + c] = f2h(e);
        }
#pragma unroll
        for (int off = 16; off; off >>= 1) sum += __shfl_xor(sum, off, 32);
        if (lane == 0) rsum[row] = 1.0f / sum;
    }
    __syncthreads();

    // ---- phase 3: ctx tile = P @ V ; wave handles 16 output cols ----
    {
        const int t = wave;
        v8f acc = {};
        for (int kc = 0; kc < 64; ++kc) {
            const v8h* pa = (const v8h*)&ps[mloc * 2048 + kc * 32 + hi * 16];
            v16h a = cat8(pa[0], pa[1]);
            union { v16h v; _Float16 e[16]; } bu;
            const float* vp = Vm + (size_t)(kc * 32 + hi * 16) * 1024 + h * 128 + t * 16 + mloc;
#pragma unroll
            for (int i = 0; i < 16; ++i) bu.e[i] = f2h(vp[(size_t)i * 1024]);
            acc = __builtin_amdgcn_wmma_f32_16x16x32_f16(false, a, false, bu.v,
                                                         (short)0, acc, false, false);
        }
#pragma unroll
        for (int r = 0; r < 8; ++r) {
            const int row = hi * 8 + r;
            ctx[(size_t)(row0 + row) * 1024 + h * 128 + t * 16 + mloc] = acc[r] * rsum[row];
        }
    }
}

// ---------------------------------------------------------------------------
// Quantizer: argmax(softmax(s/tau)) == argmax(s) (monotone), then gather-sum.
// ---------------------------------------------------------------------------
__global__ void __launch_bounds__(256)
quantize_kernel(const float* __restrict__ sims, const float* __restrict__ cb,
                float* __restrict__ quant)
{
    __shared__ float redv[8];
    __shared__ int   redi[8];
    __shared__ int   idxs[8];

    const int bn   = blockIdx.x;
    const int tid  = threadIdx.x;
    const int lane = tid & 31;
    const int wave = tid >> 5;

    for (int l = 0; l < 8; ++l) {
        float v = sims[(size_t)bn * 2048 + l * 256 + tid];
        int   bi = tid;
#pragma unroll
        for (int off = 16; off; off >>= 1) {
            const float ov = __shfl_xor(v, off, 32);
            const int   oi = __shfl_xor(bi, off, 32);
            if (ov > v || (ov == v && oi < bi)) { v = ov; bi = oi; }
        }
        if (lane == 0) { redv[wave] = v; redi[wave] = bi; }
        __syncthreads();
        if (wave == 0) {
            v  = (lane < 8) ? redv[lane] : -3.0e38f;
            bi = (lane < 8) ? redi[lane] : 0x7fffffff;
#pragma unroll
            for (int off = 16; off; off >>= 1) {
                const float ov = __shfl_xor(v, off, 32);
                const int   oi = __shfl_xor(bi, off, 32);
                if (ov > v || (ov == v && oi < bi)) { v = ov; bi = oi; }
            }
            if (lane == 0) idxs[l] = bi;
        }
        __syncthreads();
    }

    for (int d = tid; d < 1024; d += 256) {
        float s = 0.0f;
#pragma unroll
        for (int l = 0; l < 8; ++l)
            s += cb[((size_t)l * 256 + idxs[l]) * 1024 + d];
        quant[(size_t)bn * 1024 + d] = s;
    }
}

// ---------------------------------------------------------------------------
// Host-side orchestration
// ---------------------------------------------------------------------------
extern "C" void kernel_launch(void* const* d_in, const int* in_sizes, int n_in,
                              void* d_out, int out_size, void* d_ws, size_t ws_size,
                              hipStream_t stream)
{
    (void)in_sizes; (void)n_in; (void)out_size; (void)ws_size;

    const float* x    = (const float*)d_in[0];
    const float* cbk  = (const float*)d_in[1];   // [8,256,1024] == flat [2048,1024]
    const float* W1   = (const float*)d_in[2];
    const float* b1   = (const float*)d_in[3];
    const float* g1   = (const float*)d_in[4];
    const float* be1  = (const float*)d_in[5];
    const float* W2   = (const float*)d_in[6];
    const float* b2   = (const float*)d_in[7];
    const float* Wq   = (const float*)d_in[8];
    const float* bq   = (const float*)d_in[9];
    const float* Wk   = (const float*)d_in[10];
    const float* bk   = (const float*)d_in[11];
    const float* Wv   = (const float*)d_in[12];
    const float* bv   = (const float*)d_in[13];
    const float* Wo   = (const float*)d_in[14];
    const float* bo   = (const float*)d_in[15];
    float* out = (float*)d_out;

    float* ws = (float*)d_ws;
    const size_t MEG = 1024 * 1024;
    float* h1    = ws;              // 8M floats [4096,2048]
    float* enc   = ws + 8  * MEG;   // 4M [4096,1024]
    float* qb    = ws + 12 * MEG;   // 4M [4096,1024]
    float* kbuf  = ws + 16 * MEG;   // 2M [2048,1024]
    float* vbuf  = ws + 18 * MEG;   // 2M [2048,1024]
    float* ctx   = ws + 20 * MEG;   // 4M [4096,1024]   (peak 96 MB)
    float* cbo   = qb;              // reuse (q dead after attention)
    float* sims  = h1;              // reuse (h1 dead until 2nd decoder)
    float* quant = enc;             // reuse (enc dead after q projection)

    (void)hipFuncSetAttribute((const void*)attn_kernel,
                              hipFuncAttributeMaxDynamicSharedMemorySize,
                              ATTN_SMEM_BYTES);

    dim3 blk(256);
    auto gemm = [&](const float* A, const float* W, const float* bias, float* C,
                    int M, int N, int K, int wnk) {
        dim3 grid(N / 64, M / 128);
        gemm_f16_wmma<<<grid, blk, 0, stream>>>(A, W, bias, C, M, N, K, wnk);
    };

    // decoder #1
    gemm(x, W1, b1, h1, 4096, 2048, 1024, 0);
    ln_gelu_kernel<<<4096, blk, 0, stream>>>(h1, g1, be1, 2048);
    gemm(h1, W2, b2, enc, 4096, 1024, 2048, 0);

    // projections
    gemm(enc, Wq, bq, qb,   4096, 1024, 1024, 0);
    gemm(cbk, Wk, bk, kbuf, 2048, 1024, 1024, 0);
    gemm(cbk, Wv, bv, vbuf, 2048, 1024, 1024, 0);

    // attention (8 heads x 256 query tiles)
    attn_kernel<<<dim3(8, 256), blk, ATTN_SMEM_BYTES, stream>>>(qb, kbuf, vbuf, ctx);
    gemm(ctx, Wo, bo, cbo, 4096, 1024, 1024, 0);

    // codebook similarities (W stored [N,K] -> A @ W^T) + quantize
    gemm(cbo, cbk, nullptr, sims, 4096, 2048, 1024, 1);
    quantize_kernel<<<4096, blk, 0, stream>>>(sims, cbk, quant);

    // decoder #2 -> output
    gemm(quant, W1, b1, h1, 4096, 2048, 1024, 0);
    ln_gelu_kernel<<<4096, blk, 0, stream>>>(h1, g1, be1, 2048);
    gemm(h1, W2, b2, out, 4096, 1024, 2048, 0);
}
